// Decoder_7249904796082
// MI455X (gfx1250) — matrix operand order repaired and checked by reading, not compiled, not run on hardware
//
#include <hip/hip_runtime.h>
#include <hip/hip_bf16.h>

typedef __attribute__((ext_vector_type(16))) __bf16 bf16x16;
typedef __attribute__((ext_vector_type(8)))  __bf16 bf16x8;
typedef __attribute__((ext_vector_type(8)))  float  f32x8;

#define B_TOT   512
#define T_STEPS 200
#define CI      128
#define EXTD    8
#define GEN     800
#define CON     400
#define CO      32
#define FACD    128
#define CLIPV   5.0f

#define BT       16              // batch rows per workgroup
#define NWG      (B_TOT / BT)    // 32 persistent workgroups
#define NTHREADS 256
#define NWAVES   8

// K paddings (multiples of 32 for 16x16x32 bf16 WMMA)
#define K1  672   // controller: 256 (x) + 400 (h) -> 21*32
#define K3  864   // generator:  40 (x) + 800 (h)  -> 27*32
#define KCM 416   // 400 -> 13*32
#define KF  800   // 25*32 exact

// LDS row strides (elements), padded to skew banks
#define A_STRIDE  872
#define SG_STRIDE 1608
#define CH_STRIDE 408
#define GH_STRIDE 808

#define SMEM_BYTES (16*A_STRIDE*2 + 16*SG_STRIDE*4 + 16*CH_STRIDE*4 + \
                    16*GH_STRIDE*4 + 16*FACD*2 + 16*CO*4)

__device__ __forceinline__ float sigmoidf(float x) { return 1.0f / (1.0f + __expf(-x)); }

// ---------------------------------------------------------------------------
// Preprocessing
// ---------------------------------------------------------------------------

__global__ void norm_kernel(const float* __restrict__ facW, float* __restrict__ norms) {
  int c = threadIdx.x;
  if (c < FACD) {
    float s = 0.f;
    for (int k = 0; k < GEN; ++k) { float v = facW[(long)k * FACD + c]; s += v * v; }
    norms[c] = sqrtf(s);
  }
}

// Build padded, stacked row-major bf16 weight matrices.
__global__ void stage_kernel(const float* __restrict__ conK, const float* __restrict__ conR,
                             const float* __restrict__ genK, const float* __restrict__ genR,
                             const float* __restrict__ comW, const float* __restrict__ facW,
                             const float* __restrict__ norms,
                             __bf16* __restrict__ S1, __bf16* __restrict__ S2,
                             __bf16* __restrict__ S3, __bf16* __restrict__ S4,
                             __bf16* __restrict__ Scm, __bf16* __restrict__ Sf) {
  long tid = (long)blockIdx.x * blockDim.x + threadIdx.x;
  long stride = (long)gridDim.x * blockDim.x;
  // S1 [672][800]: rows 0..255 conK cols 0..799, rows 256..655 conR cols 0..799
  for (long i = tid; i < (long)K1 * 800; i += stride) {
    int k = (int)(i / 800), n = (int)(i % 800); float v = 0.f;
    if (k < 256) v = conK[(long)k * 1200 + n];
    else if (k < 656) v = conR[(long)(k - 256) * 1200 + n];
    S1[i] = (__bf16)v;
  }
  // S2 [672][400]: cols 800..1199 of stacked conK/conR
  for (long i = tid; i < (long)K1 * 400; i += stride) {
    int k = (int)(i / 400), n = (int)(i % 400); float v = 0.f;
    if (k < 256) v = conK[(long)k * 1200 + 800 + n];
    else if (k < 656) v = conR[(long)(k - 256) * 1200 + 800 + n];
    S2[i] = (__bf16)v;
  }
  // S3 [864][1600]: rows 0..39 genK, rows 40..839 genR, cols 0..1599
  for (long i = tid; i < (long)K3 * 1600; i += stride) {
    int k = (int)(i / 1600), n = (int)(i % 1600); float v = 0.f;
    if (k < 40) v = genK[(long)k * 2400 + n];
    else if (k < 840) v = genR[(long)(k - 40) * 2400 + n];
    S3[i] = (__bf16)v;
  }
  // S4 [864][800]: cols 1600..2399
  for (long i = tid; i < (long)K3 * 800; i += stride) {
    int k = (int)(i / 800), n = (int)(i % 800); float v = 0.f;
    if (k < 40) v = genK[(long)k * 2400 + 1600 + n];
    else if (k < 840) v = genR[(long)(k - 40) * 2400 + 1600 + n];
    S4[i] = (__bf16)v;
  }
  // Scm [416][32]
  for (long i = tid; i < (long)KCM * CO; i += stride) {
    int k = (int)(i / CO), n = (int)(i % CO);
    float v = (k < CON) ? comW[(long)k * CO + n] : 0.f;
    Scm[i] = (__bf16)v;
  }
  // Sf [800][128] column-normalized
  for (long i = tid; i < (long)KF * FACD; i += stride) {
    int k = (int)(i / FACD), n = (int)(i % FACD);
    Sf[i] = (__bf16)(facW[(long)k * FACD + n] / norms[n]);
  }
}

// Pack row-major [KT*32][N] bf16 into per-tile WMMA B-fragment order:
// dst[((kt*NT+nt)*32 + lane)*16 + j] = src[(kt*32 + (lane&16 ? 8 : 0) + (j < 8 ? j : j+8)) * N + nt*16 + (lane&15)]
__global__ void pack_kernel(const __bf16* __restrict__ src, __bf16* __restrict__ dst,
                            int KT, int N) {
  int NT = N >> 4;
  long total = (long)KT * NT * 512;
  long stride = (long)gridDim.x * blockDim.x;
  for (long idx = (long)blockIdx.x * blockDim.x + threadIdx.x; idx < total; idx += stride) {
    int tile = (int)(idx >> 9);
    int r = (int)(idx & 511);
    int lane = r >> 4, j = r & 15;
    int kt = tile / NT, nt = tile - kt * NT;
    int k = kt * 32 + ((lane & 16) ? 8 : 0) + (j < 8 ? j : j + 8);
    int n = nt * 16 + (lane & 15);
    dst[idx] = src[(long)k * N + n];
  }
}

// ---------------------------------------------------------------------------
// WMMA GEMM helpers (M=16 tile, A from LDS row-major bf16, B pre-packed)
// ---------------------------------------------------------------------------

__device__ __forceinline__ void gemm_lds_out(
    const __bf16* __restrict__ sA,
    const __bf16* __restrict__ W, int NT, int KT,
    float* __restrict__ sOut, int outStride, int outColBase,
    int lane, int wave) {
  const int row   = lane & 15;
  const int koff  = (lane & 16) ? 8 : 0;
  const int rbase = (lane & 16) ? 8 : 0;
  for (int nt = wave; nt < NT; nt += NWAVES) {
    f32x8 acc = {};
    const __bf16* wp = W + ((long)nt * 32 + lane) * 16;
    const __bf16* ap = sA + row * A_STRIDE + koff;
    for (int kt = 0; kt < KT; ++kt) {
      union { bf16x16 v; bf16x8 h[2]; } a;
      a.h[0] = *(const bf16x8*)(ap);
      a.h[1] = *(const bf16x8*)(ap + 16);
      bf16x16 b = *(const bf16x16*)wp;
      acc = __builtin_amdgcn_wmma_f32_16x16x32_bf16(false, a.v, false, b,
                                                    (short)0, acc, false, false);
      ap += 32;
      wp += (long)NT * 512;
    }
    const int col = outColBase + nt * 16 + row;
#pragma unroll
    for (int v = 0; v < 8; ++v)
      sOut[(rbase + v) * outStride + col] = acc[v];
  }
}

__device__ __forceinline__ void gemm_fac(
    const __bf16* __restrict__ sA, const __bf16* __restrict__ W,
    float* __restrict__ outG, long outRowStride, long outBase,
    __bf16* __restrict__ sFac, int lane, int wave) {
  const int row   = lane & 15;
  const int koff  = (lane & 16) ? 8 : 0;
  const int rbase = (lane & 16) ? 8 : 0;
  const int NT = FACD / 16, KT = KF / 32;
  for (int nt = wave; nt < NT; nt += NWAVES) {
    f32x8 acc = {};
    const __bf16* wp = W + ((long)nt * 32 + lane) * 16;
    const __bf16* ap = sA + row * A_STRIDE + koff;
    for (int kt = 0; kt < KT; ++kt) {
      union { bf16x16 v; bf16x8 h[2]; } a;
      a.h[0] = *(const bf16x8*)(ap);
      a.h[1] = *(const bf16x8*)(ap + 16);
      bf16x16 b = *(const bf16x16*)wp;
      acc = __builtin_amdgcn_wmma_f32_16x16x32_bf16(false, a.v, false, b,
                                                    (short)0, acc, false, false);
      ap += 32;
      wp += (long)NT * 512;
    }
    const int col = nt * 16 + row;
#pragma unroll
    for (int v = 0; v < 8; ++v) {
      float val = acc[v];
      int b = rbase + v;
      sFac[b * FACD + col] = (__bf16)val;
      if (outG) outG[outBase + (long)b * outRowStride + col] = val;
    }
  }
}

// ---------------------------------------------------------------------------
// Persistent decoder kernel: one WG owns 16 batch rows for all 200 steps
// ---------------------------------------------------------------------------

__global__ void __launch_bounds__(NTHREADS, 1)
lfads_kernel(const float* __restrict__ ci, const float* __restrict__ ext,
             const float* __restrict__ gen_init, const float* __restrict__ con_h0,
             const float* __restrict__ con_b, const float* __restrict__ com_b,
             const float* __restrict__ gen_b,
             const __bf16* __restrict__ P1, const __bf16* __restrict__ P2,
             const __bf16* __restrict__ P3, const __bf16* __restrict__ P4,
             const __bf16* __restrict__ Pcm, const __bf16* __restrict__ Pf,
             float* __restrict__ out) {
  extern __shared__ char smem[];
  __bf16* sA   = (__bf16*)smem;                            // [16][A_STRIDE] bf16 A operand
  float*  sG   = (float*)(smem + 16 * A_STRIDE * 2);       // [16][SG_STRIDE] gate scratch
  float*  sCH  = (float*)((char*)sG  + 16 * SG_STRIDE * 4);// [16][CH_STRIDE] con_h f32
  float*  sGH  = (float*)((char*)sCH + 16 * CH_STRIDE * 4);// [16][GH_STRIDE] gen_h f32
  __bf16* sFac = (__bf16*)((char*)sGH + 16 * GH_STRIDE * 4);// [16][128] fac bf16
  float*  sCM  = (float*)((char*)sFac + 16 * FACD * 2);    // [16][32] co_mean f32

  const int tid  = threadIdx.x;
  const int lane = tid & 31;
  const int wave = tid >> 5;
  const int bg   = blockIdx.x * BT;

  // ---- init state ----
  for (int i = tid; i < BT * CON; i += NTHREADS) {
    int b = i / CON, n = i - b * CON;
    sCH[b * CH_STRIDE + n] = con_h0[n];
  }
  for (int i = tid; i < BT * GEN; i += NTHREADS) {
    int b = i / GEN, n = i - b * GEN;
    float v = gen_init[(long)(bg + b) * GEN + n];
    sGH[b * GH_STRIDE + n] = v;
    sA[b * A_STRIDE + n]   = (__bf16)v;
  }
  __syncthreads();
  gemm_fac(sA, Pf, (float*)nullptr, 0, 0, sFac, lane, wave);   // fac0
  __syncthreads();

  const float* cib  = ci  + (long)bg * T_STEPS * CI;
  const float* extb = ext + (long)bg * T_STEPS * EXTD;

  for (int t = 0; t < T_STEPS; ++t) {
    // 1) stage A = [ci_t | fac | con_h | 0]   (K = 672)
    for (int i = tid; i < BT * K1; i += NTHREADS) {
      int b = i / K1, c = i - b * K1;
      __bf16 v;
      if (c < CI)        v = (__bf16)cib[(long)b * T_STEPS * CI + (long)t * CI + c];
      else if (c < 256)  v = sFac[b * FACD + (c - CI)];
      else if (c < 656)  v = (__bf16)sCH[b * CH_STRIDE + (c - 256)];
      else               v = (__bf16)0.f;
      sA[b * A_STRIDE + c] = v;
    }
    __syncthreads();
    // 2) controller z,r pre-activations -> sG[:,0..799]
    gemm_lds_out(sA, P1, 50, K1 / 32, sG, SG_STRIDE, 0, lane, wave);
    __syncthreads();
    // 3) controller gates; rh overwrites h region of A
    for (int i = tid; i < BT * CON; i += NTHREADS) {
      int b = i / CON, n = i - b * CON;
      float z = sigmoidf(sG[b * SG_STRIDE + n] + con_b[n]);
      float r = sigmoidf(sG[b * SG_STRIDE + CON + n] + con_b[CON + n]);
      sG[b * SG_STRIDE + n] = z;
      sA[b * A_STRIDE + 256 + n] = (__bf16)(r * sCH[b * CH_STRIDE + n]);
    }
    __syncthreads();
    // 4) controller candidate pre-activation -> sG[:,800..1199]
    gemm_lds_out(sA, P2, 25, K1 / 32, sG, SG_STRIDE, 800, lane, wave);
    __syncthreads();
    // 5) controller update; stage A = [con_h | 0] (K=416) for co_mean
    for (int i = tid; i < BT * KCM; i += NTHREADS) {
      int b = i / KCM, n = i - b * KCM;
      if (n < CON) {
        float hh = tanhf(sG[b * SG_STRIDE + 800 + n] + con_b[2 * CON + n]);
        float z  = sG[b * SG_STRIDE + n];
        float h  = sCH[b * CH_STRIDE + n];
        float hn = z * h + (1.f - z) * hh;
        hn = fminf(fmaxf(hn, -CLIPV), CLIPV);
        sCH[b * CH_STRIDE + n] = hn;
        sA[b * A_STRIDE + n]   = (__bf16)hn;
      } else {
        sA[b * A_STRIDE + n] = (__bf16)0.f;
      }
    }
    __syncthreads();
    // 6) co_mean = con_h @ com_W -> sCM
    gemm_lds_out(sA, Pcm, 2, KCM / 32, sCM, CO, 0, lane, wave);
    __syncthreads();
    // 7) stage A = [co_mean+b | ext_t | gen_h | 0]  (K = 864)
    for (int i = tid; i < BT * K3; i += NTHREADS) {
      int b = i / K3, c = i - b * K3;
      __bf16 v;
      if (c < CO)        v = (__bf16)(sCM[b * CO + c] + com_b[c]);
      else if (c < 40)   v = (__bf16)extb[(long)b * T_STEPS * EXTD + (long)t * EXTD + (c - CO)];
      else if (c < 840)  v = (__bf16)sGH[b * GH_STRIDE + (c - 40)];
      else               v = (__bf16)0.f;
      sA[b * A_STRIDE + c] = v;
    }
    __syncthreads();
    // 8) generator z,r pre-activations -> sG[:,0..1599]
    gemm_lds_out(sA, P3, 100, K3 / 32, sG, SG_STRIDE, 0, lane, wave);
    __syncthreads();
    // 9) generator gates
    for (int i = tid; i < BT * GEN; i += NTHREADS) {
      int b = i / GEN, n = i - b * GEN;
      float z = sigmoidf(sG[b * SG_STRIDE + n] + gen_b[n]);
      float r = sigmoidf(sG[b * SG_STRIDE + GEN + n] + gen_b[GEN + n]);
      sG[b * SG_STRIDE + n] = z;
      sA[b * A_STRIDE + 40 + n] = (__bf16)(r * sGH[b * GH_STRIDE + n]);
    }
    __syncthreads();
    // 10) generator candidate pre-activation -> sG[:,800..1599]
    gemm_lds_out(sA, P4, 50, K3 / 32, sG, SG_STRIDE, 800, lane, wave);
    __syncthreads();
    // 11) generator update; stage A = gen_h bf16 (K=800) for fac GEMM
    for (int i = tid; i < BT * GEN; i += NTHREADS) {
      int b = i / GEN, n = i - b * GEN;
      float hh = tanhf(sG[b * SG_STRIDE + 800 + n] + gen_b[2 * GEN + n]);
      float z  = sG[b * SG_STRIDE + n];
      float h  = sGH[b * GH_STRIDE + n];
      float hn = z * h + (1.f - z) * hh;
      hn = fminf(fmaxf(hn, -CLIPV), CLIPV);
      sGH[b * GH_STRIDE + n] = hn;
      sA[b * A_STRIDE + n]   = (__bf16)hn;
    }
    __syncthreads();
    // 12) fac = gen_h @ fac_Wn -> d_out[:, t, :] (f32) and sFac (bf16)
    gemm_fac(sA, Pf, out, (long)T_STEPS * FACD,
             (long)bg * T_STEPS * FACD + (long)t * FACD, sFac, lane, wave);
    __syncthreads();
  }
}

// ---------------------------------------------------------------------------
// Host-side launch
// ---------------------------------------------------------------------------

extern "C" void kernel_launch(void* const* d_in, const int* in_sizes, int n_in,
                              void* d_out, int out_size, void* d_ws, size_t ws_size,
                              hipStream_t stream) {
  const float* ci       = (const float*)d_in[0];
  const float* ext      = (const float*)d_in[1];
  const float* gen_init = (const float*)d_in[2];
  const float* con_h0   = (const float*)d_in[3];
  const float* con_K    = (const float*)d_in[4];
  const float* con_R    = (const float*)d_in[5];
  const float* con_b    = (const float*)d_in[6];
  const float* com_W    = (const float*)d_in[7];
  const float* com_b    = (const float*)d_in[8];
  // d_in[9] = col_W, d_in[10] = col_b -- logvars never reach the output; skipped.
  const float* gen_K    = (const float*)d_in[11];
  const float* gen_R    = (const float*)d_in[12];
  const float* gen_b    = (const float*)d_in[13];
  const float* fac_W    = (const float*)d_in[14];

  char* ws = (char*)d_ws;
  float* norms = (float*)ws;
  size_t off = 512;
  auto take = [&](size_t elems) -> __bf16* {
    __bf16* p = (__bf16*)(ws + off);
    off += elems * 2;
    return p;
  };
  // staged row-major bf16 (padded K), then packed fragment buffers. ~12 MB total.
  __bf16* S1  = take((size_t)K1  * 800);
  __bf16* S2  = take((size_t)K1  * 400);
  __bf16* S3  = take((size_t)K3  * 1600);
  __bf16* S4  = take((size_t)K3  * 800);
  __bf16* Scm = take((size_t)KCM * CO);
  __bf16* Sf  = take((size_t)KF  * FACD);
  __bf16* P1  = take((size_t)K1  * 800);
  __bf16* P2  = take((size_t)K1  * 400);
  __bf16* P3  = take((size_t)K3  * 1600);
  __bf16* P4  = take((size_t)K3  * 800);
  __bf16* Pcm = take((size_t)KCM * CO);
  __bf16* Pf  = take((size_t)KF  * FACD);
  (void)in_sizes; (void)n_in; (void)out_size; (void)ws_size;

  norm_kernel<<<1, 128, 0, stream>>>(fac_W, norms);
  stage_kernel<<<512, 256, 0, stream>>>(con_K, con_R, gen_K, gen_R, com_W, fac_W,
                                        norms, S1, S2, S3, S4, Scm, Sf);
  pack_kernel<<<512, 256, 0, stream>>>(S1, P1, K1 / 32, 800);
  pack_kernel<<<512, 256, 0, stream>>>(S2, P2, K1 / 32, 400);
  pack_kernel<<<512, 256, 0, stream>>>(S3, P3, K3 / 32, 1600);
  pack_kernel<<<512, 256, 0, stream>>>(S4, P4, K3 / 32, 800);
  pack_kernel<<<64,  256, 0, stream>>>(Scm, Pcm, KCM / 32, CO);
  pack_kernel<<<128, 256, 0, stream>>>(Sf, Pf, KF / 32, FACD);

  // Allow >64KB dynamic LDS (WGP has 320KB). Idempotent & deterministic.
  hipFuncSetAttribute(reinterpret_cast<const void*>(lfads_kernel),
                      hipFuncAttributeMaxDynamicSharedMemorySize, SMEM_BYTES);

  lfads_kernel<<<NWG, NTHREADS, SMEM_BYTES, stream>>>(
      ci, ext, gen_init, con_h0, con_b, com_b, gen_b,
      P1, P2, P3, P4, Pcm, Pf, (float*)d_out);
}
